// RNN_33011118637314
// MI455X (gfx1250) — matrix operand order repaired and checked by reading, not compiled, not run on hardware
//
#include <hip/hip_runtime.h>

// ---------------------------------------------------------------------------
// Linear RNN  h_t = A h_{t-1} + B x_t ,  y_t = C h_t + D*x_t
// L=8192, d_model=512, d_hidden=1024.
// Chunked two-pass scan (NC=128 chunks of T=64): every time step is a dense
// [1024x1024]x[1024x128] matmul on v_wmma_f32_16x16x32_bf16 (bf16 operands,
// f32 accumulation). Inter-chunk carries via A^64 (6 WMMA squarings) + 127
// tiny f32 matvec launches. GEMMs: 4 waves/block, 4 M-tiles/wave; the shared
// 16-row B-strip is staged into LDS by the Tensor Data Mover.
// ---------------------------------------------------------------------------

constexpr int L_  = 8192;
constexpr int DM  = 512;
constexpr int DH  = 1024;
constexpr int CT  = 64;        // time steps per chunk
constexpr int NC  = L_ / CT;   // 128 chunks

typedef __attribute__((ext_vector_type(16))) __bf16 v16bf;
typedef __attribute__((ext_vector_type(8)))  __bf16 v8bf;
typedef __attribute__((ext_vector_type(8)))  float  v8f;
typedef unsigned int u32x4 __attribute__((ext_vector_type(4)));
typedef int          i32x4 __attribute__((ext_vector_type(4)));
typedef int          i32x8 __attribute__((ext_vector_type(8)));

union BF8 { v8bf v; unsigned short u[8]; };

#define DEV __device__ __forceinline__

DEV unsigned short f2bf(float f) {
  union { float f; unsigned int u; } x; x.f = f;
  unsigned int r = x.u + 0x7FFFu + ((x.u >> 16) & 1u);   // round-to-nearest-even
  return (unsigned short)(r >> 16);
}

// ----------------------- Tensor Data Mover staging -------------------------
// 2D tile load: rows x cols bf16 elements, arbitrary row stride, into LDS.
// D# per CDNA5 ISA 8.3/8.4: group0 = {flags, lds_addr, global_addr, type=2},
// group1 = {data_size=2B, tensor_dim0/1, tile_dim0/1, dim0_stride}.
#if __has_builtin(__builtin_amdgcn_tensor_load_to_lds) && \
    __has_builtin(__builtin_amdgcn_s_wait_tensorcnt)
#define HAVE_TDM 1
DEV void tdm_load_2d(unsigned lds_byte_off, const void* gptr,
                     unsigned rows, unsigned cols,
                     unsigned long long row_stride_elems) {
  unsigned long long ga = (unsigned long long)gptr;
  u32x4 g0;
  g0[0] = 1u;                                            // count=1 (valid), user mode
  g0[1] = lds_byte_off;                                  // lds_addr (bytes)
  g0[2] = (unsigned)ga;                                  // global_addr[31:0]
  g0[3] = (unsigned)((ga >> 32) & 0x01FFFFFFull) | (2u << 30); // addr[56:32]|type=2
  i32x8 g1;
  g1[0] = (int)(1u << 16);                               // data_size=1 -> 2 bytes
  g1[1] = (int)((cols & 0xFFFFu) << 16);                 // tensor_dim0[15:0]
  g1[2] = (int)(((cols >> 16) & 0xFFFFu) |               // tensor_dim0[31:16]
                ((rows & 0xFFFFu) << 16));               // tensor_dim1[15:0]
  g1[3] = (int)(((rows >> 16) & 0xFFFFu) |               // tensor_dim1[31:16]
                ((cols & 0xFFFFu) << 16));               // tile_dim0
  g1[4] = (int)(rows & 0xFFFFu);                         // tile_dim1 (tile_dim2=0)
  g1[5] = (int)(unsigned)(row_stride_elems);             // dim0_stride[31:0]
  g1[6] = (int)(unsigned)((row_stride_elems >> 32) & 0xFFFFull); // dim0_stride[47:32]
  g1[7] = 0;
  i32x4 z4 = {0, 0, 0, 0};
#if defined(__clang_major__) && (__clang_major__ >= 23)
  i32x8 z8 = {0, 0, 0, 0, 0, 0, 0, 0};
  __builtin_amdgcn_tensor_load_to_lds(g0, g1, z4, z4, z8, 0);
#else
  __builtin_amdgcn_tensor_load_to_lds(g0, g1, z4, z4, 0);
#endif
}
#endif

// Stage 16 B-rows (K elems each, strided) into LDS; all 128 threads arrive.
template <int K>
DEV void stage_B(unsigned short* sB, const unsigned short* __restrict__ Bstrip,
                 unsigned long long row_stride_elems) {
#ifdef HAVE_TDM
  if ((threadIdx.x >> 5) == 0) {            // wave 0 drives the TDM
    tdm_load_2d((unsigned)(unsigned long long)(void*)sB, Bstrip, 16u, (unsigned)K,
                row_stride_elems);
    __builtin_amdgcn_s_wait_tensorcnt(0);
  }
#else
  for (int c = threadIdx.x; c < 16 * K / 8; c += 128) {
    int r = c / (K / 8), col = c % (K / 8);
    *(v8bf*)(sB + r * K + col * 8) =
        *(const v8bf*)(Bstrip + (size_t)r * row_stride_elems + col * 8);
  }
#endif
  __syncthreads();
}

// ------------------------- 4-tile NT WMMA core -----------------------------
// Wave computes D[64m x 16n]: A rows from global (row-major, stride K),
// B-strip from LDS. acc layout: acc[mi][v] = D[m0+mi*16+mh*8+v][n0+(lane&15)].
template <int K>
DEV void gemm4_ldsB(const unsigned short* __restrict__ Aop, int m0_wave,
                    const unsigned short* __restrict__ Bstrip,
                    unsigned long long brow_stride, v8f acc[4]) {
  __shared__ unsigned short sB[16 * K];
  stage_B<K>(sB, Bstrip, brow_stride);
  const int lane = threadIdx.x & 31;
  const int kh = lane >> 4;
  const unsigned short* ap = Aop + (size_t)(m0_wave + (lane & 15)) * K + kh * 8;
  const unsigned short* bp = sB + (size_t)(lane & 15) * K + kh * 8;
  __builtin_prefetch(ap, 0, 3);             // global_prefetch_b8
#pragma unroll 2
  for (int k = 0; k < K; k += 32) {
    v8bf b0 = *(const v8bf*)(bp + k);       // ds_load_b128
    v8bf b1 = *(const v8bf*)(bp + k + 16);
    v16bf b;
#pragma unroll
    for (int i = 0; i < 8; ++i) { b[i] = b0[i]; b[i + 8] = b1[i]; }
#pragma unroll
    for (int mi = 0; mi < 4; ++mi) {
      const unsigned short* a = ap + (size_t)(mi * 16) * K;
      v8bf a0 = *(const v8bf*)(a + k);
      v8bf a1 = *(const v8bf*)(a + k + 16);
      v16bf av;
#pragma unroll
      for (int i = 0; i < 8; ++i) { av[i] = a0[i]; av[i + 8] = a1[i]; }
      acc[mi] = __builtin_amdgcn_wmma_f32_16x16x32_bf16(false, av, false, b, (short)0,
                                                        acc[mi], false, false);
    }
  }
}

// --------------------------- small utility kernels -------------------------

__global__ void k_cvt(const float* __restrict__ s, unsigned short* __restrict__ d, int n) {
  int i = blockIdx.x * 256 + threadIdx.x;
  if (i < n) d[i] = f2bf(s[i]);
}

__global__ void k_cvtT(const float* __restrict__ s, unsigned short* __restrict__ d,
                       int rows, int cols) {   // d[c][r] = bf16(s[r][c])
  int i = blockIdx.x * 256 + threadIdx.x;
  if (i < rows * cols) {
    int r = i / cols, c = i % cols;
    d[(size_t)c * rows + r] = f2bf(s[(size_t)r * cols + c]);
  }
}

__global__ void k_zero(float* p, int n) {
  int i = blockIdx.x * 256 + threadIdx.x;
  if (i < n) p[i] = 0.0f;
}

// Seed pass-1 t=0:  S_0 = Bu[c*CT]
__global__ void k_seed(const float* __restrict__ Bu, unsigned short* __restrict__ HS) {
  int i = blockIdx.x * 256 + threadIdx.x;      // NC*DH threads
  int c = i >> 10, h = i & (DH - 1);
  HS[((size_t)c * CT) * DH + h] = f2bf(Bu[((size_t)c * CT) * DH + h]);
}

// Inter-chunk scan step (f32): hout[m] = r[m] + sum_k A64T[k][m] * hin[k]
__global__ void k_scan(const float* __restrict__ A64T, const float* __restrict__ r,
                       const float* __restrict__ hin, float* __restrict__ hout) {
  int m = blockIdx.x * 256 + threadIdx.x;      // 1024 threads
  float acc = r[m];
  for (int k = 0; k < DH; ++k) acc = fmaf(A64T[(size_t)k * DH + m], hin[k], acc);
  hout[m] = acc;
}

// ------------------------------ WMMA kernels -------------------------------

// Bu[l][h] = sum_d B[h][d] * X[l][d]   (M=h=1024 -> 4 mblocks, N=l=8192)
__global__ void __launch_bounds__(128)
k_gemm_bu(const unsigned short* __restrict__ Bbf, const unsigned short* __restrict__ Xbf,
          float* __restrict__ Bu) {
  const int mb = blockIdx.x & 3, nb = blockIdx.x >> 2;
  const int lane = threadIdx.x & 31, mh = lane >> 4;
  const int m0 = mb * 256 + (threadIdx.x >> 5) * 64;
  const int n0 = nb * 16, nl = n0 + (lane & 15);
  v8f acc[4] = {};
  gemm4_ldsB<DM>(Bbf, m0, Xbf + (size_t)n0 * DM, DM, acc);
#pragma unroll
  for (int mi = 0; mi < 4; ++mi)
    *(v8f*)(Bu + (size_t)nl * DH + m0 + mi * 16 + mh * 8) = acc[mi];
}

// Q = P @ P  (B-operand from PT); emits Q (row-major), QT, optional QT f32.
__global__ void __launch_bounds__(128)
k_gemm_sq(const unsigned short* __restrict__ P, const unsigned short* __restrict__ PT,
          unsigned short* __restrict__ Q, unsigned short* __restrict__ QT,
          float* __restrict__ QT_f32) {
  const int mb = blockIdx.x & 3, nb = blockIdx.x >> 2;   // 4 x 64 blocks
  const int lane = threadIdx.x & 31, mh = lane >> 4;
  const int m0 = mb * 256 + (threadIdx.x >> 5) * 64;
  const int nl = nb * 16 + (lane & 15);
  v8f acc[4] = {};
  gemm4_ldsB<DH>(P, m0, PT + (size_t)(nb * 16) * DH, DH, acc);
#pragma unroll
  for (int mi = 0; mi < 4; ++mi) {
    BF8 q;
#pragma unroll
    for (int v = 0; v < 8; ++v) q.u[v] = f2bf(acc[mi][v]);
    const int m = m0 + mi * 16 + mh * 8;
    *(v8bf*)(QT + (size_t)nl * DH + m) = q.v;
#pragma unroll
    for (int v = 0; v < 8; ++v) Q[(size_t)(m + v) * DH + nl] = q.u[v];
    if (QT_f32) *(v8f*)(QT_f32 + (size_t)nl * DH + m) = acc[mi];
  }
}

// One recurrence step over all chunks:
//   S_t[h][c] = sum_k A[h][k]*Sprev[row(c)][k] + Bu[c*CT+t][h]
// Sprev row(c) = Sprev + (c*sStride + sOff)*DH ; out -> HS[c*CT+t][h] (bf16).
__global__ void __launch_bounds__(128)
k_recur(const unsigned short* __restrict__ Abf,
        const unsigned short* __restrict__ Sprev, int sStride, int sOff,
        const float* __restrict__ Bu, unsigned short* __restrict__ HS, int t,
        float* __restrict__ rOut) {
  const int mb = blockIdx.x & 3, nb = blockIdx.x >> 2;   // 4 x 8 blocks
  const int lane = threadIdx.x & 31, mh = lane >> 4;
  const int m0 = mb * 256 + (threadIdx.x >> 5) * 64;
  const int n0 = nb * 16, nl = n0 + (lane & 15);
  v8f acc[4];
#pragma unroll
  for (int mi = 0; mi < 4; ++mi)
    acc[mi] = *(const v8f*)(Bu + ((size_t)nl * CT + t) * DH + m0 + mi * 16 + mh * 8);
  gemm4_ldsB<DH>(Abf, m0,
                 Sprev + ((size_t)n0 * sStride + sOff) * DH,
                 (unsigned long long)sStride * DH, acc);
#pragma unroll
  for (int mi = 0; mi < 4; ++mi) {
    BF8 q;
#pragma unroll
    for (int v = 0; v < 8; ++v) q.u[v] = f2bf(acc[mi][v]);
    const int m = m0 + mi * 16 + mh * 8;
    *(v8bf*)(HS + ((size_t)nl * CT + t) * DH + m) = q.v;
    if (rOut) *(v8f*)(rOut + (size_t)nl * DH + m) = acc[mi];  // f32 carry
  }
}

// Y[l][dm] = sum_h C[dm][h]*HS[l][h] + Dv[dm]*X[l][dm]  (M=dm=512 -> 2 mblocks)
__global__ void __launch_bounds__(128)
k_gemm_out(const unsigned short* __restrict__ Cbf, const unsigned short* __restrict__ HS,
           const float* __restrict__ X, const float* __restrict__ Dv,
           float* __restrict__ Y) {
  const int mb = blockIdx.x & 1, nb = blockIdx.x >> 1;   // 2 x 512 blocks
  const int lane = threadIdx.x & 31, mh = lane >> 4;
  const int m0 = mb * 256 + (threadIdx.x >> 5) * 64;
  const int nl = nb * 16 + (lane & 15);
  v8f acc[4] = {};
  gemm4_ldsB<DH>(Cbf, m0, HS + (size_t)(nb * 16) * DH, DH, acc);
#pragma unroll
  for (int mi = 0; mi < 4; ++mi) {
    const int m = m0 + mi * 16 + mh * 8;
    v8f d = *(const v8f*)(Dv + m);
    v8f x = *(const v8f*)(X + (size_t)nl * DM + m);
#pragma unroll
    for (int v = 0; v < 8; ++v) acc[mi][v] += d[v] * x[v];
    *(v8f*)(Y + (size_t)nl * DM + m) = acc[mi];
  }
}

// ------------------------------- launcher ----------------------------------

static inline char* carve(char*& p, size_t bytes) {
  char* r = p;
  p += (bytes + 255) & ~(size_t)255;
  return r;
}

extern "C" void kernel_launch(void* const* d_in, const int* in_sizes, int n_in,
                              void* d_out, int out_size, void* d_ws, size_t ws_size,
                              hipStream_t stream) {
  const float* X  = (const float*)d_in[0];   // [L, DM]
  const float* A  = (const float*)d_in[1];   // [DH, DH]
  const float* Bm = (const float*)d_in[2];   // [DH, DM]
  const float* Cm = (const float*)d_in[3];   // [DM, DH]
  const float* Dv = (const float*)d_in[4];   // [DM]
  float* Y = (float*)d_out;                  // [L, DM]

  char* w = (char*)d_ws;
  unsigned short* Xbf  = (unsigned short*)carve(w, (size_t)L_ * DM * 2);
  unsigned short* Abf  = (unsigned short*)carve(w, (size_t)DH * DH * 2);
  unsigned short* ATbf = (unsigned short*)carve(w, (size_t)DH * DH * 2);
  unsigned short* Bbf  = (unsigned short*)carve(w, (size_t)DH * DM * 2);
  unsigned short* Cbf  = (unsigned short*)carve(w, (size_t)DM * DH * 2);
  unsigned short* HS   = (unsigned short*)carve(w, (size_t)L_ * DH * 2);
  unsigned short* pw0  = (unsigned short*)carve(w, (size_t)DH * DH * 2);
  unsigned short* pw0T = (unsigned short*)carve(w, (size_t)DH * DH * 2);
  unsigned short* pw1  = (unsigned short*)carve(w, (size_t)DH * DH * 2);
  unsigned short* pw1T = (unsigned short*)carve(w, (size_t)DH * DH * 2);
  unsigned short* h0bf = (unsigned short*)carve(w, (size_t)NC * DH * 2);
  float* Bu   = (float*)carve(w, (size_t)L_ * DH * 4);
  float* A64T = (float*)carve(w, (size_t)DH * DH * 4);
  float* rbuf = (float*)carve(w, (size_t)NC * DH * 4);
  float* h0   = (float*)carve(w, (size_t)NC * DH * 4);

  // 1) bf16 copies (A also transposed for the squaring chain's B-operand)
  k_cvt <<<(L_ * DM + 255) / 256, 256, 0, stream>>>(X,  Xbf, L_ * DM);
  k_cvt <<<(DH * DH + 255) / 256, 256, 0, stream>>>(A,  Abf, DH * DH);
  k_cvt <<<(DH * DM + 255) / 256, 256, 0, stream>>>(Bm, Bbf, DH * DM);
  k_cvt <<<(DM * DH + 255) / 256, 256, 0, stream>>>(Cm, Cbf, DM * DH);
  k_cvtT<<<(DH * DH + 255) / 256, 256, 0, stream>>>(A, ATbf, DH, DH);

  // 2) Bu = X @ B^T  (stored [l][h], f32)
  k_gemm_bu<<<4 * (L_ / 16), 128, 0, stream>>>(Bbf, Xbf, Bu);

  // 3) A^64 via 6 squarings (last also emits f32 transposed copy for the scan)
  {
    const unsigned short* P = Abf; const unsigned short* PT = ATbf;
    for (int s = 0; s < 6; ++s) {
      unsigned short* Q  = (s & 1) ? pw1  : pw0;
      unsigned short* QT = (s & 1) ? pw1T : pw0T;
      k_gemm_sq<<<4 * (DH / 16), 128, 0, stream>>>(P, PT, Q, QT,
                                                   (s == 5) ? A64T : nullptr);
      P = Q; PT = QT;
    }
  }

  // 4) Pass 1: zero-init chunk recurrence -> carries r_c = S_{CT-1}
  k_seed<<<(NC * DH) / 256, 256, 0, stream>>>(Bu, HS);
  for (int t = 1; t < CT; ++t)
    k_recur<<<4 * (NC / 16), 128, 0, stream>>>(Abf, HS, CT, t - 1, Bu, HS, t,
                                               (t == CT - 1) ? rbuf : nullptr);

  // 5) Inter-chunk scan in f32: h0_{c+1} = A^64 h0_c + r_c
  k_zero<<<(DH + 255) / 256, 256, 0, stream>>>(h0, DH);
  for (int c = 0; c < NC - 1; ++c)
    k_scan<<<DH / 256, 256, 0, stream>>>(A64T, rbuf + (size_t)c * DH,
                                         h0 + (size_t)c * DH, h0 + (size_t)(c + 1) * DH);
  k_cvt<<<(NC * DH + 255) / 256, 256, 0, stream>>>(h0, h0bf, NC * DH);

  // 6) Pass 2: recurrence with true initial states, stores all hidden states
  k_recur<<<4 * (NC / 16), 128, 0, stream>>>(Abf, h0bf, 1, 0, Bu, HS, 0, nullptr);
  for (int t = 1; t < CT; ++t)
    k_recur<<<4 * (NC / 16), 128, 0, stream>>>(Abf, HS, CT, t - 1, Bu, HS, t, nullptr);

  // 7) Y = H @ C^T + D * X
  k_gemm_out<<<2 * (L_ / 16), 128, 0, stream>>>(Cbf, HS, X, Dv, Y);

  (void)in_sizes; (void)n_in; (void)out_size; (void)ws_size;
}
